// QNN_27290222198801
// MI455X (gfx1250) — matrix operand order repaired and checked by reading, not compiled, not run on hardware
//
#include <hip/hip_runtime.h>
#include <math.h>

typedef __attribute__((ext_vector_type(16))) _Float16 v16h;
typedef __attribute__((ext_vector_type(8)))  _Float16 v8h;
typedef __attribute__((ext_vector_type(8)))  float    v8f;

#define NQ     10
#define NL     4
#define DIM    1024          // 2^NQ
#define INF    64            // IN_FEATURES
#define QBATCH 32768
#define NTILE  (DIM / 16)    // 64
#define KTILE  (DIM / 32)    // 32
#define ROWS   64            // samples per block in main kernel
#define RS     1032          // padded phi row stride (halfs): lanes land 4 banks apart

// ---------------------------------------------------------------------------
// Kernel 1: angles = tanh(x @ W_pre^T + b_pre); store cos/sin of half-angles.
// ---------------------------------------------------------------------------
__global__ void qnn_angles(const float* __restrict__ x,
                           const float* __restrict__ Wpre,
                           const float* __restrict__ bpre,
                           float* __restrict__ cs /* [BATCH][NQ][2] */) {
  __shared__ float w[NQ * INF];
  for (int i = threadIdx.x; i < NQ * INF; i += blockDim.x) w[i] = Wpre[i];
  __syncthreads();

  int s = blockIdx.x * blockDim.x + threadIdx.x;
  const float* xr = x + (size_t)s * INF;
  float xv[INF];
#pragma unroll
  for (int i = 0; i < INF; i += 4) {
    float4 t = *(const float4*)(xr + i);
    xv[i] = t.x; xv[i + 1] = t.y; xv[i + 2] = t.z; xv[i + 3] = t.w;
  }
#pragma unroll
  for (int q = 0; q < NQ; ++q) {
    float acc = bpre[q];
#pragma unroll
    for (int i = 0; i < INF; ++i) acc += xv[i] * w[q * INF + i];
    float a = tanhf(acc) * 0.5f;
    float cc, ss;
    __sincosf(a, &ss, &cc);
    cs[((size_t)s * NQ + q) * 2 + 0] = cc;
    cs[((size_t)s * NQ + q) * 2 + 1] = ss;
  }
}

// ---------------------------------------------------------------------------
// Kernel 2: simulate the entangling layers on basis column t = blockIdx.x,
// giving U[s,t] for all s. Emit fp16 fragment packings for the M-build GEMM:
//   A-layout (M=t, K=s):  aRe/aIm[(mT*32+kT)*512 + lane*16 + h] = w_s * U[s,t]
//   B-layout (N=t, K=s):  bRe/bIm[(nT*32+kT)*512 + lane*16 + h] =        U[s,t]
// A-layout: lanes 0-15 -> M=lane, K in {0..7,16..23}; lanes 16-31 -> {8..15,24..31}
// B-layout: lane = N%16 + 16*((Kloc>>3)&1), h = Kloc%8 + 8*(Kloc>=16)
// ---------------------------------------------------------------------------
__global__ void qnn_unitary(const float* __restrict__ qw, // [NL][NQ][3]
                            const float* __restrict__ Wpost,
                            _Float16* __restrict__ aRe, _Float16* __restrict__ aIm,
                            _Float16* __restrict__ bRe, _Float16* __restrict__ bIm) {
  __shared__ float2 st[DIM];
  const int col = blockIdx.x;
  const int tid = threadIdx.x;          // 128 threads
  for (int i = tid; i < DIM; i += blockDim.x)
    st[i] = make_float2(i == col ? 1.f : 0.f, 0.f);
  __syncthreads();

  for (int l = 0; l < NL; ++l) {
    for (int q = 0; q < NQ; ++q) {
      float phi = qw[(l * NQ + q) * 3 + 0];
      float th  = qw[(l * NQ + q) * 3 + 1];
      float om  = qw[(l * NQ + q) * 3 + 2];
      float sth, cth; __sincosf(th * 0.5f, &sth, &cth);
      float sp, cp;   __sincosf((phi + om) * 0.5f, &sp, &cp);
      float sm, cm;   __sincosf((phi - om) * 0.5f, &sm, &cm);
      float2 m00 = { cp * cth, -sp * cth };
      float2 m01 = { -cm * sth, -sm * sth };
      float2 m10 = { cm * sth, -sm * sth };
      float2 m11 = { cp * cth,  sp * cth };
      int bitpos = 9 - q;
      int stride = 1 << bitpos;
      for (int p = tid; p < DIM / 2; p += blockDim.x) {
        int i0 = ((p >> bitpos) << (bitpos + 1)) | (p & (stride - 1));
        int i1 = i0 | stride;
        float2 a = st[i0], b = st[i1];
        float2 r0 = { m00.x * a.x - m00.y * a.y + m01.x * b.x - m01.y * b.y,
                      m00.x * a.y + m00.y * a.x + m01.x * b.y + m01.y * b.x };
        float2 r1 = { m10.x * a.x - m10.y * a.y + m11.x * b.x - m11.y * b.y,
                      m10.x * a.y + m10.y * a.x + m11.x * b.y + m11.y * b.x };
        st[i0] = r0; st[i1] = r1;
      }
      __syncthreads();
    }
    int r = (l % (NQ - 1)) + 1;
    for (int q = 0; q < NQ; ++q) {
      int c = q, t = (q + r) % NQ;
      int cpos = 9 - c, tpos = 9 - t;
      int cmask = 1 << cpos, tmask = 1 << tpos;
      for (int p = tid; p < DIM / 4; p += blockDim.x) {
        int i = 0, pb = p;
        for (int b = 0; b < NQ; ++b) {
          if (b == cpos || b == tpos) continue;
          i |= (pb & 1) << b;
          pb >>= 1;
        }
        i |= cmask;
        int j = i | tmask;
        float2 tmp = st[i]; st[i] = st[j]; st[j] = tmp;
      }
      __syncthreads();
    }
  }

  float wp[NQ];
#pragma unroll
  for (int q = 0; q < NQ; ++q) wp[q] = Wpost[q];

  const int mT = col >> 4, mLoc = col & 15;
  for (int s = tid; s < DIM; s += blockDim.x) {
    float wsc = 0.f;
#pragma unroll
    for (int q = 0; q < NQ; ++q) wsc += wp[q] * (((s >> (9 - q)) & 1) ? -1.f : 1.f);
    float2 v = st[s];
    int kT = s >> 5, kl = s & 31;
    // A-layout position
    int laneA = mLoc + (((kl >> 3) & 1) << 4);
    int hA = (kl & 7) | (((kl >> 4) & 1) << 3);
    size_t offA = ((size_t)(mT * KTILE + kT) * 32 + laneA) * 16 + hA;
    aRe[offA] = (_Float16)(wsc * v.x);
    aIm[offA] = (_Float16)(wsc * v.y);
    // B-layout position
    int laneB = mLoc + (((kl >> 3) & 1) << 4);
    int hB = (kl & 7) | (((kl >> 4) & 1) << 3);
    size_t offB = ((size_t)(mT * KTILE + kT) * 32 + laneB) * 16 + hB;
    bRe[offB] = (_Float16)v.x;
    bIm[offB] = (_Float16)v.y;
  }
}

// ---------------------------------------------------------------------------
// Kernel 3: M = A_re x B_re + A_im x B_im  (1024x1024 real symmetric),
// one 16x16 tile per wave, K=1024 via 32 x WMMA pairs. Result scattered in
// fp16 into the main-GEMM B-fragment layout (using M's symmetry).
// ---------------------------------------------------------------------------
__global__ void __launch_bounds__(128)
qnn_mbuild(const _Float16* __restrict__ aRe, const _Float16* __restrict__ aIm,
           const _Float16* __restrict__ bRe, const _Float16* __restrict__ bIm,
           _Float16* __restrict__ Mp) {
  const int lane = threadIdx.x & 31, wave = threadIdx.x >> 5;
  const int tile = blockIdx.x * 4 + wave;
  const int iT = tile >> 6, jT = tile & 63;
  const v16h* AR = (const v16h*)aRe; const v16h* AI = (const v16h*)aIm;
  const v16h* BR = (const v16h*)bRe; const v16h* BI = (const v16h*)bIm;
  v8f acc = {};
#pragma unroll 4
  for (int k = 0; k < KTILE; ++k) {
    v16h ar = AR[((size_t)iT * KTILE + k) * 32 + lane];
    v16h br = BR[((size_t)jT * KTILE + k) * 32 + lane];
    v16h ai = AI[((size_t)iT * KTILE + k) * 32 + lane];
    v16h bi = BI[((size_t)jT * KTILE + k) * 32 + lane];
    acc = __builtin_amdgcn_wmma_f32_16x16x32_f16(false, ar, false, br, (short)0, acc, false, false);
    acc = __builtin_amdgcn_wmma_f32_16x16x32_f16(false, ai, false, bi, (short)0, acc, false, false);
  }
  // C/D layout: VGPR j8, lanes 0-15 -> row i=j8, lanes 16-31 -> i=j8+8; col j=lane%16
#pragma unroll
  for (int j8 = 0; j8 < 8; ++j8) {
    int i = iT * 16 + j8 + ((lane >> 4) << 3);
    int j = jT * 16 + (lane & 15);
    // main-GEMM B layout with (N = i, K = j) using M[i,j] = M[j,i]
    size_t off = ((size_t)((i >> 4) * KTILE + (j >> 5)) * 32 +
                  ((i & 15) + (((j >> 3) & 1) << 4))) * 16 + ((j & 7) | (((j >> 4) & 1) << 3));
    Mp[off] = (_Float16)acc[j8];
  }
}

// ---------------------------------------------------------------------------
// Kernel 4: main fused GEMM  out[b] = phi_b^T M phi_b + b_post.
// 128 threads = 4 waves, 64 samples/block. phi staged in LDS (padded rows).
// 4M x 4N register blocking: per K-step 4 ds_load + 4 global fragment loads
// feed 16 v_wmma. Epilogue fuses the phi.y dot via LDS gathers; cross-wave
// partials merged with ds_add_f32 atomics.
// ---------------------------------------------------------------------------
__global__ void __launch_bounds__(128)
qnn_main(const float* __restrict__ cs,
         const _Float16* __restrict__ Mp,
         const float* __restrict__ bpost,
         float* __restrict__ out) {
  extern __shared__ _Float16 smem[];
  _Float16* phi = smem;                        // [ROWS][RS]
  float* outSum = (float*)(smem + ROWS * RS);  // [ROWS]

  const int tid = threadIdx.x;
  const int lane = tid & 31, wave = tid >> 5;
  const int blockBase = blockIdx.x * ROWS;

  if (tid < ROWS) outSum[tid] = 0.f;

  // ---- stage phi: thread owns row tid>>1, k_tiles [(tid&1)*16, +16)
  {
    int row = tid >> 1;
    int ktBase = (tid & 1) * 16;
    int sample = blockBase + row;
    float cq[NQ], sq[NQ];
#pragma unroll
    for (int q = 0; q < NQ; ++q) {
      cq[q] = cs[((size_t)sample * NQ + q) * 2 + 0];
      sq[q] = cs[((size_t)sample * NQ + q) * 2 + 1];
    }
    float suf[32];
#pragma unroll
    for (int kl = 0; kl < 32; ++kl) {
      float v = 1.f;
#pragma unroll
      for (int b = 0; b < 5; ++b) v *= ((kl >> b) & 1) ? sq[9 - b] : cq[9 - b];
      suf[kl] = v;
    }
    for (int kt = ktBase; kt < ktBase + 16; ++kt) {
      float pre = 1.f;
#pragma unroll
      for (int b = 0; b < 5; ++b) pre *= ((kt >> b) & 1) ? sq[4 - b] : cq[4 - b];
      v16h v0, v1;
#pragma unroll
      for (int kl = 0; kl < 16; ++kl) v0[kl] = (_Float16)(pre * suf[kl]);
#pragma unroll
      for (int kl = 0; kl < 16; ++kl) v1[kl] = (_Float16)(pre * suf[16 + kl]);
      *(v16h*)&phi[row * RS + kt * 32] = v0;
      *(v16h*)&phi[row * RS + kt * 32 + 16] = v1;
    }
  }
  __syncthreads();

  const v16h* MB = (const v16h*)Mp;
  const int o1 = (lane & 16) ? 8 : 0;   // A-layout K-offset for this half-wave
  float psum[4][8];
#pragma unroll
  for (int mt = 0; mt < 4; ++mt)
#pragma unroll
    for (int j = 0; j < 8; ++j) psum[mt][j] = 0.f;

  for (int quad = 0; quad < 4; ++quad) {
    const int n0 = (wave * 4 + quad) * 4;   // this wave's 4 n_tiles
    v8f acc[4][4];
#pragma unroll
    for (int mt = 0; mt < 4; ++mt)
#pragma unroll
      for (int nn = 0; nn < 4; ++nn) acc[mt][nn] = (v8f){};

    if (quad + 1 < 4) {
      __builtin_prefetch(&MB[((size_t)(n0 + 4) * KTILE) * 32 + lane], 0, 3);
    }

#pragma unroll 2
    for (int kt = 0; kt < KTILE; ++kt) {
      v16h A[4];
#pragma unroll
      for (int mt = 0; mt < 4; ++mt) {
        const _Float16* pr = &phi[(mt * 16 + (lane & 15)) * RS + kt * 32 + o1];
        v8h lo = *(const v8h*)pr;
        v8h hi = *(const v8h*)(pr + 16);
        A[mt] = __builtin_shufflevector(lo, hi, 0, 1, 2, 3, 4, 5, 6, 7,
                                        8, 9, 10, 11, 12, 13, 14, 15);
      }
#pragma unroll
      for (int nn = 0; nn < 4; ++nn) {
        v16h B = MB[((size_t)(n0 + nn) * KTILE + kt) * 32 + lane];
#pragma unroll
        for (int mt = 0; mt < 4; ++mt)
          acc[mt][nn] = __builtin_amdgcn_wmma_f32_16x16x32_f16(
              false, A[mt], false, B, (short)0, acc[mt][nn], false, false);
      }
    }

    // epilogue: psum[row] += phi[row][s] * y[row][s]
#pragma unroll
    for (int nn = 0; nn < 4; ++nn) {
      int s = (n0 + nn) * 16 + (lane & 15);
#pragma unroll
      for (int mt = 0; mt < 4; ++mt)
#pragma unroll
        for (int j = 0; j < 8; ++j) {
          int row = mt * 16 + j + ((lane >> 4) << 3);
          psum[mt][j] += (float)phi[row * RS + s] * acc[mt][nn][j];
        }
    }
  }

  // reduce each row over its 16 lanes
#pragma unroll
  for (int off = 1; off < 16; off <<= 1)
#pragma unroll
    for (int mt = 0; mt < 4; ++mt)
#pragma unroll
      for (int j = 0; j < 8; ++j) psum[mt][j] += __shfl_xor(psum[mt][j], off, 32);

  if ((lane & 15) == 0) {
#pragma unroll
    for (int mt = 0; mt < 4; ++mt)
#pragma unroll
      for (int j = 0; j < 8; ++j)
        atomicAdd(&outSum[mt * 16 + j + ((lane >> 4) << 3)], psum[mt][j]);
  }
  __syncthreads();
  if (tid < ROWS) out[blockBase + tid] = outSum[tid] + bpost[0];
}

// ---------------------------------------------------------------------------
extern "C" void kernel_launch(void* const* d_in, const int* in_sizes, int n_in,
                              void* d_out, int out_size, void* d_ws, size_t ws_size,
                              hipStream_t stream) {
  const float* x     = (const float*)d_in[0];
  const float* Wpre  = (const float*)d_in[1];
  const float* bpre  = (const float*)d_in[2];
  const float* qw    = (const float*)d_in[3];
  const float* Wpost = (const float*)d_in[4];
  const float* bpost = (const float*)d_in[5];
  float* out = (float*)d_out;

  // Workspace: cs | aRe | aIm | bRe | bIm | Mp
  char* ws = (char*)d_ws;
  size_t off = 0;
  float* cs = (float*)(ws + off);
  off += ((size_t)QBATCH * NQ * 2 * sizeof(float) + 255) & ~(size_t)255;
  const size_t mat = (size_t)DIM * DIM * sizeof(_Float16);
  _Float16* aRe = (_Float16*)(ws + off); off += mat;
  _Float16* aIm = (_Float16*)(ws + off); off += mat;
  _Float16* bRe = (_Float16*)(ws + off); off += mat;
  _Float16* bIm = (_Float16*)(ws + off); off += mat;
  _Float16* Mp  = (_Float16*)(ws + off); off += mat;

  const size_t smemMain = (size_t)ROWS * RS * sizeof(_Float16) + ROWS * sizeof(float);
  hipFuncSetAttribute((const void*)qnn_main,
                      hipFuncAttributeMaxDynamicSharedMemorySize, (int)smemMain);

  qnn_angles <<<QBATCH / 256, 256, 0, stream>>>(x, Wpre, bpre, cs);
  qnn_unitary<<<DIM, 128, 0, stream>>>(qw, Wpost, aRe, aIm, bRe, bIm);
  qnn_mbuild <<<DIM, 128, 0, stream>>>(aRe, aIm, bRe, bIm, Mp);
  qnn_main   <<<QBATCH / ROWS, 128, smemMain, stream>>>(cs, Mp, bpost, out);
}
